// DeepseekV3Attention_89876485636733
// MI455X (gfx1250) — matrix operand order repaired and checked by reading, not compiled, not run on hardware
//
#include <hip/hip_runtime.h>
#include <hip/hip_bf16.h>

typedef __attribute__((ext_vector_type(16))) __bf16 v16bf;
typedef __attribute__((ext_vector_type(8)))  __bf16 v8bf;
typedef __attribute__((ext_vector_type(8)))  float  v8f;

#define B_SZ 2
#define S_LEN 1024
#define HID 4096
#define NH 32
#define NOPE 128
#define ROPE 64
#define QKH 192
#define VH 128
#define QLR 1536
#define KVLR 512
#define MROWS (B_SZ * S_LEN)   // 2048

__device__ inline v8f zero8() {
  v8f z;
#pragma unroll
  for (int i = 0; i < 8; ++i) z[i] = 0.f;
  return z;
}

// A-fragment (16x32 bf16, MxK): lane holds row M=lane%16.
// elements 0..15 -> K = 16*half + 0..15 (contiguous)
__device__ inline v16bf load_a16(const __bf16* base, int half) {
  v8bf a0 = *(const v8bf*)(base + 16 * half);
  v8bf a1 = *(const v8bf*)(base + 16 * half + 8);
  v16bf r;
#pragma unroll
  for (int i = 0; i < 8; ++i) { r[i] = a0[i]; r[i + 8] = a1[i]; }
  return r;
}

// B-fragment (32x16 bf16, KxN): lane holds column N=lane%16,
// K = 16*half + e, e=0..15 contiguous -> single 32B load.
__device__ inline v16bf load_b16(const __bf16* base, int half) {
  return *(const v16bf*)(base + 16 * half);
}

// Generic (flat) pointer to LDS: hardware uses addr[31:0] as the LDS address.
__device__ inline unsigned lds_addr32(const void* p) {
  return (unsigned)(size_t)p;
}

// Async DMA: global -> LDS, 16 bytes per active lane (tracked by ASYNCcnt).
__device__ inline void async_g2l_b128(unsigned lds_off, const void* gaddr) {
  asm volatile("global_load_async_to_lds_b128 %0, %1, off"
               :: "v"(lds_off), "v"((unsigned long long)(size_t)gaddr)
               : "memory");
}

__device__ inline void wait_asynccnt0() {
  asm volatile("s_wait_asynccnt 0x0" ::: "memory");
}

// ---------------------------------------------------------------------------
// fp32 -> bf16 flat convert
__global__ __launch_bounds__(256) void cvt_bf16_kernel(const float* __restrict__ x,
                                                       __bf16* __restrict__ y, size_t n) {
  size_t i = (size_t)blockIdx.x * 256 + threadIdx.x;
  size_t stride = (size_t)gridDim.x * 256;
  for (; i < n; i += stride) y[i] = (__bf16)x[i];
}

// fp32 W[K,N] -> bf16 Wt[N,K] (tiled transpose)
__global__ __launch_bounds__(256) void transpose_bf16_kernel(const float* __restrict__ W,
                                                             __bf16* __restrict__ Wt,
                                                             int K, int N) {
  __shared__ float tile[32][33];
  const int nb = blockIdx.x * 32;
  const int kb = blockIdx.y * 32;
  const int tx = threadIdx.x & 31;
  const int ty = threadIdx.x >> 5;  // 0..7
#pragma unroll
  for (int i = 0; i < 32; i += 8) {
    int k = kb + ty + i, n = nb + tx;
    if (k < K && n < N) tile[ty + i][tx] = W[(size_t)k * N + n];
  }
  __syncthreads();
#pragma unroll
  for (int i = 0; i < 32; i += 8) {
    int n = nb + ty + i, k = kb + tx;
    if (n < N && k < K) Wt[(size_t)n * K + k] = (__bf16)tile[tx][ty + i];
  }
}

// ---------------------------------------------------------------------------
// C[M,N] (f32) = A[M,K] (bf16, row-major) * Wt[N,K] (bf16, row-major = B^T)
// block = 128 threads (4 waves); wave w -> rows [64*bx + 16w, +16), cols [64*by, +64)
// B tile (64 cols x 32 k = 4KB) staged in LDS by async DMA, double buffered.
__global__ __launch_bounds__(128) void gemm_bf16_kernel(const __bf16* __restrict__ A,
                                                        const __bf16* __restrict__ Wt,
                                                        float* __restrict__ C,
                                                        int M, int N, int K) {
  __shared__ __bf16 bT[2][64 * 32];   // [buf][n_local * 32 + k_local]
  const int tid = threadIdx.x;
  const int wave = tid >> 5;
  const int lane = tid & 31;
  const int half = lane >> 4;
  const int lr = lane & 15;
  const int m0 = blockIdx.x * 64 + wave * 16;
  const int n0 = blockIdx.y * 64;
  const int row = m0 + lr;

  // Each thread DMAs two b128 chunks of the 4KB tile (256 chunks total).
  auto issue_btile = [&](int k0, int buf) {
#pragma unroll
    for (int i = 0; i < 2; ++i) {
      const int chunk = tid + 128 * i;     // 0..255
      const int rowN = chunk >> 2;         // 0..63
      const int cc = chunk & 3;            // 4 chunks of 8 bf16 per row
      const __bf16* g = Wt + (size_t)(n0 + rowN) * K + k0 + 8 * cc;
      async_g2l_b128(lds_addr32(&bT[buf][rowN * 32 + 8 * cc]), g);
    }
  };

  v8f acc[4];
#pragma unroll
  for (int t = 0; t < 4; ++t) acc[t] = zero8();

  const __bf16* arow = A + (size_t)row * K;
  issue_btile(0, 0);

  for (int k0 = 0, it = 0; k0 < K; k0 += 32, ++it) {
    const int buf = it & 1;
    wait_asynccnt0();      // previous tile's DMA complete (this wave's share)
    __syncthreads();       // all waves' shares visible
    if (k0 + 32 < K) issue_btile(k0 + 32, buf ^ 1);   // prefetch next tile

    if (k0 + 32 < K) __builtin_prefetch(arow + k0 + 32, 0, 1);
    v16bf a = load_a16(arow + k0, half);
#pragma unroll
    for (int t = 0; t < 4; ++t) {
      v16bf b = load_b16(&bT[buf][(16 * t + lr) * 32], half);
      acc[t] = __builtin_amdgcn_wmma_f32_16x16x32_bf16(false, a, false, b,
                                                       (short)0, acc[t], false, false);
    }
    __syncthreads();       // all waves done reading buf before it is re-filled
  }
  // C layout: element r -> M = m0 + r + 8*half, N = n0 + 16*t + lr
#pragma unroll
  for (int t = 0; t < 4; ++t)
#pragma unroll
    for (int r = 0; r < 8; ++r)
      C[(size_t)(m0 + r + 8 * half) * N + (n0 + 16 * t + lr)] = acc[t][r];
}

// ---------------------------------------------------------------------------
// RMSNorm over N columns of fp32 X (leading dim ldx), times w, to bf16 Y (ld ldy)
__global__ __launch_bounds__(256) void rmsnorm_bf16_kernel(const float* __restrict__ X, int ldx,
                                                           int N, const float* __restrict__ w,
                                                           __bf16* __restrict__ Y, int ldy) {
  __shared__ float red[256];
  const int row = blockIdx.x;
  const int tid = threadIdx.x;
  const float* x = X + (size_t)row * ldx;
  float s = 0.f;
  for (int i = tid; i < N; i += 256) { float v = x[i]; s += v * v; }
  red[tid] = s;
  __syncthreads();
  for (int st = 128; st > 0; st >>= 1) {
    if (tid < st) red[tid] += red[tid + st];
    __syncthreads();
  }
  const float inv = rsqrtf(red[0] / (float)N + 1e-6f);
  for (int i = tid; i < N; i += 256)
    Y[(size_t)row * ldy + i] = (__bf16)(x[i] * inv * w[i]);
}

// ---------------------------------------------------------------------------
// qf fp32 [M, H*192] -> Qm bf16 [B,H,S,192] with RoPE on last 64 dims of each head
__global__ __launch_bounds__(256) void build_q_kernel(const float* __restrict__ qf,
                                                      const float* __restrict__ cosb,
                                                      const float* __restrict__ sinb,
                                                      __bf16* __restrict__ Qm) {
  const int m = blockIdx.x;          // b*S + s
  const int b = m >> 10;
  const int s = m & (S_LEN - 1);
  const float* qrow = qf + (size_t)m * (NH * QKH);
  const float* c = cosb + (size_t)m * ROPE;
  const float* sn = sinb + (size_t)m * ROPE;
  for (int idx = threadIdx.x; idx < NH * QKH; idx += 256) {
    const int h = idx / QKH, d = idx % QKH;
    float v;
    if (d < NOPE) {
      v = qrow[h * QKH + d];
    } else {
      const int dr = d - NOPE;
      const float x = qrow[h * QKH + NOPE + dr];
      const float xh = (dr < 32) ? -qrow[h * QKH + NOPE + dr + 32]
                                 :  qrow[h * QKH + NOPE + dr - 32];
      v = x * c[dr] + xh * sn[dr];
    }
    Qm[((size_t)(b * NH + h) * S_LEN + s) * QKH + d] = (__bf16)v;
  }
}

// kvbf fp32 [M, H*256] -> Km bf16 [B,H,S,192] (k_pass + broadcast roped k_rot),
//                         Vt bf16 [B,H,128,S]
__global__ __launch_bounds__(256) void build_kv_kernel(const float* __restrict__ kvbf,
                                                       const float* __restrict__ kvf,
                                                       const float* __restrict__ cosb,
                                                       const float* __restrict__ sinb,
                                                       __bf16* __restrict__ Km,
                                                       __bf16* __restrict__ Vt) {
  const int m = blockIdx.x;
  const int b = m >> 10;
  const int s = m & (S_LEN - 1);
  const float* row = kvbf + (size_t)m * (NH * (NOPE + VH));
  const float* kv = kvf + (size_t)m * (KVLR + ROPE);
  const float* c = cosb + (size_t)m * ROPE;
  const float* sn = sinb + (size_t)m * ROPE;
  for (int idx = threadIdx.x; idx < NH * (NOPE + VH); idx += 256) {
    const int h = idx >> 8, d = idx & 255;
    const float v = row[idx];
    if (d < NOPE)
      Km[((size_t)(b * NH + h) * S_LEN + s) * QKH + d] = (__bf16)v;
    else
      Vt[((size_t)(b * NH + h) * VH + (d - NOPE)) * S_LEN + s] = (__bf16)v;
  }
  for (int idx = threadIdx.x; idx < NH * ROPE; idx += 256) {
    const int h = idx >> 6, dr = idx & 63;
    const float x = kv[KVLR + dr];
    const float xh = (dr < 32) ? -kv[KVLR + dr + 32] : kv[KVLR + dr - 32];
    const float v = x * c[dr] + xh * sn[dr];
    Km[((size_t)(b * NH + h) * S_LEN + s) * QKH + NOPE + dr] = (__bf16)v;
  }
}

// ---------------------------------------------------------------------------
// Flash-style causal attention. 1 wave per (b, h, 16-query tile).
// Out bf16 [B*S, H*VH] (head-major columns, matching o_w row order).
__global__ __launch_bounds__(32) void mla_attn_kernel(const __bf16* __restrict__ Qm,
                                                      const __bf16* __restrict__ Km,
                                                      const __bf16* __restrict__ Vt,
                                                      __bf16* __restrict__ Out) {
  __shared__ __bf16 pL[16 * 32];
  const int lane = threadIdx.x & 31;
  const int half = lane >> 4;
  const int lr = lane & 15;
  const int qb = blockIdx.x & 63;          // S/16 tiles
  const int h = (blockIdx.x >> 6) & 31;
  const int b = blockIdx.x >> 11;
  const int q0 = qb * 16;

  const __bf16* Q = Qm + (size_t)(b * NH + h) * S_LEN * QKH;
  const __bf16* Kp = Km + (size_t)(b * NH + h) * S_LEN * QKH;
  const __bf16* V = Vt + (size_t)(b * NH + h) * VH * S_LEN;

  // Q fragments for this 16-row tile (d = 6 chunks of 32)
  v16bf qa[6];
  {
    const __bf16* qrow = Q + (size_t)(q0 + lr) * QKH;
#pragma unroll
    for (int c = 0; c < 6; ++c) qa[c] = load_a16(qrow + 32 * c, half);
  }

  float rm[8], rl[8];
  v8f o[8];
#pragma unroll
  for (int r = 0; r < 8; ++r) { rm[r] = -3.0e38f; rl[r] = 0.f; }
#pragma unroll
  for (int t = 0; t < 8; ++t) o[t] = zero8();

  const float scale = 0.07216878364870323f;  // 192^-0.5

  for (int k0 = 0; k0 < q0 + 16; k0 += 32) {
    // scores: two 16-key subtiles
    v8f s0 = zero8(), s1 = zero8();
#pragma unroll
    for (int c = 0; c < 6; ++c) {
      v16bf b0 = load_b16(Kp + (size_t)(k0 + lr) * QKH + 32 * c, half);
      s0 = __builtin_amdgcn_wmma_f32_16x16x32_bf16(false, qa[c], false, b0,
                                                   (short)0, s0, false, false);
      v16bf b1 = load_b16(Kp + (size_t)(k0 + 16 + lr) * QKH + 32 * c, half);
      s1 = __builtin_amdgcn_wmma_f32_16x16x32_bf16(false, qa[c], false, b1,
                                                   (short)0, s1, false, false);
    }
    // online softmax; row m = q0 + r + 8*half lives in the 16 lanes of this half
#pragma unroll
    for (int r = 0; r < 8; ++r) {
      const int qi = q0 + r + 8 * half;
      float v0 = s0[r] * scale + (((k0 + lr) <= qi) ? 0.f : -1.0e9f);
      float v1 = s1[r] * scale + (((k0 + 16 + lr) <= qi) ? 0.f : -1.0e9f);
      float mx = fmaxf(v0, v1);
#pragma unroll
      for (int msk = 1; msk < 16; msk <<= 1) mx = fmaxf(mx, __shfl_xor(mx, msk, 32));
      const float mnew = fmaxf(rm[r], mx);
      const float alpha = __expf(rm[r] - mnew);
      const float p0 = __expf(v0 - mnew);
      const float p1 = __expf(v1 - mnew);
      float ps = p0 + p1;
#pragma unroll
      for (int msk = 1; msk < 16; msk <<= 1) ps += __shfl_xor(ps, msk, 32);
      rl[r] = rl[r] * alpha + ps;
      rm[r] = mnew;
#pragma unroll
      for (int t = 0; t < 8; ++t) o[t][r] *= alpha;
      pL[(r + 8 * half) * 32 + lr] = (__bf16)p0;
      pL[(r + 8 * half) * 32 + 16 + lr] = (__bf16)p1;
    }
    // same-wave DS ops are in-order: no barrier needed for store->load
    v16bf pa = load_a16(&pL[lr * 32], half);
#pragma unroll
    for (int t = 0; t < 8; ++t) {
      v16bf vb = load_b16(V + (size_t)(16 * t + lr) * S_LEN + k0, half);
      o[t] = __builtin_amdgcn_wmma_f32_16x16x32_bf16(false, pa, false, vb,
                                                     (short)0, o[t], false, false);
    }
  }
  // finalize: divide by l, write bf16 [B*S, H*VH]
#pragma unroll
  for (int t = 0; t < 8; ++t) {
#pragma unroll
    for (int r = 0; r < 8; ++r) {
      const float val = o[t][r] / rl[r];
      const size_t rowo = (size_t)b * S_LEN + q0 + r + 8 * half;
      Out[rowo * (NH * VH) + h * VH + 16 * t + lr] = (__bf16)val;
    }
  }
}

// ---------------------------------------------------------------------------
extern "C" void kernel_launch(void* const* d_in, const int* in_sizes, int n_in,
                              void* d_out, int out_size, void* d_ws, size_t ws_size,
                              hipStream_t stream) {
  const float* hidden = (const float*)d_in[0];
  const float* cosb = (const float*)d_in[1];
  const float* sinb = (const float*)d_in[2];
  // d_in[3] = attention_mask (pure causal; recomputed in-kernel)
  const float* q_a_w = (const float*)d_in[4];
  const float* q_a_ln_w = (const float*)d_in[5];
  const float* q_b_w = (const float*)d_in[6];
  const float* kv_a_w = (const float*)d_in[7];
  const float* kv_a_ln_w = (const float*)d_in[8];
  const float* kv_b_w = (const float*)d_in[9];
  const float* o_w = (const float*)d_in[10];
  float* out = (float*)d_out;

  char* ws = (char*)d_ws;
  size_t off = 0;
  auto alloc = [&](size_t bytes) -> void* {
    void* p = ws + off;
    off = (off + bytes + 255) & ~(size_t)255;
    return p;
  };

  __bf16* hB    = (__bf16*)alloc((size_t)MROWS * HID * 2);
  __bf16* WqaT  = (__bf16*)alloc((size_t)QLR * HID * 2);
  __bf16* WqbT  = (__bf16*)alloc((size_t)NH * QKH * QLR * 2);
  __bf16* WkvaT = (__bf16*)alloc((size_t)(KVLR + ROPE) * HID * 2);
  __bf16* WkvbT = (__bf16*)alloc((size_t)NH * (NOPE + VH) * KVLR * 2);
  __bf16* WoT   = (__bf16*)alloc((size_t)HID * (NH * VH) * 2);
  float*  cqF   = (float*)alloc((size_t)MROWS * QLR * 4);
  float*  kvF   = (float*)alloc((size_t)MROWS * (KVLR + ROPE) * 4);
  __bf16* cqB   = (__bf16*)alloc((size_t)MROWS * QLR * 2);
  __bf16* ckvB  = (__bf16*)alloc((size_t)MROWS * KVLR * 2);
  float*  qF    = (float*)alloc((size_t)MROWS * NH * QKH * 4);
  float*  kvbF  = (float*)alloc((size_t)MROWS * NH * (NOPE + VH) * 4);
  __bf16* Qm    = (__bf16*)alloc((size_t)B_SZ * NH * S_LEN * QKH * 2);
  __bf16* Km    = (__bf16*)alloc((size_t)B_SZ * NH * S_LEN * QKH * 2);
  __bf16* Vtm   = (__bf16*)alloc((size_t)B_SZ * NH * VH * S_LEN * 2);
  __bf16* attnB = (__bf16*)alloc((size_t)MROWS * NH * VH * 2);

  // 1) hidden -> bf16
  {
    size_t n = (size_t)MROWS * HID;
    cvt_bf16_kernel<<<dim3((unsigned)((n + 255) / 256)), 256, 0, stream>>>(hidden, hB, n);
  }
  // 2) weight transposes (fp32 [K,N] -> bf16 [N,K])
  auto xpose = [&](const float* W, __bf16* Wt, int K, int N) {
    transpose_bf16_kernel<<<dim3((N + 31) / 32, (K + 31) / 32), 256, 0, stream>>>(W, Wt, K, N);
  };
  xpose(q_a_w, WqaT, HID, QLR);
  xpose(q_b_w, WqbT, QLR, NH * QKH);
  xpose(kv_a_w, WkvaT, HID, KVLR + ROPE);
  xpose(kv_b_w, WkvbT, KVLR, NH * (NOPE + VH));
  xpose(o_w, WoT, NH * VH, HID);

  auto gemm = [&](const __bf16* A, const __bf16* Wt, float* C, int M, int N, int K) {
    gemm_bf16_kernel<<<dim3(M / 64, N / 64), 128, 0, stream>>>(A, Wt, C, M, N, K);
  };

  // 3) low-rank projections
  gemm(hB, WqaT, cqF, MROWS, QLR, HID);
  gemm(hB, WkvaT, kvF, MROWS, KVLR + ROPE, HID);

  // 4) RMSNorms
  rmsnorm_bf16_kernel<<<MROWS, 256, 0, stream>>>(cqF, QLR, QLR, q_a_ln_w, cqB, QLR);
  rmsnorm_bf16_kernel<<<MROWS, 256, 0, stream>>>(kvF, KVLR + ROPE, KVLR, kv_a_ln_w, ckvB, KVLR);

  // 5) up projections
  gemm(cqB, WqbT, qF, MROWS, NH * QKH, QLR);
  gemm(ckvB, WkvbT, kvbF, MROWS, NH * (NOPE + VH), KVLR);

  // 6) RoPE + layout packing
  build_q_kernel<<<MROWS, 256, 0, stream>>>(qF, cosb, sinb, Qm);
  build_kv_kernel<<<MROWS, 256, 0, stream>>>(kvbF, kvF, cosb, sinb, Km, Vtm);

  // 7) attention core: B*H*(S/16) single-wave blocks
  mla_attn_kernel<<<B_SZ * NH * (S_LEN / 16), 32, 0, stream>>>(Qm, Km, Vtm, attnB);

  // 8) output projection -> fp32 d_out
  gemm(attnB, WoT, out, MROWS, HID, NH * VH);
}
